// SparseResUNet_75814762709626
// MI455X (gfx1250) — matrix operand order repaired and checked by reading, not compiled, not run on hardware
//
#include <hip/hip_runtime.h>
#include <hip/hip_bf16.h>

typedef __attribute__((ext_vector_type(16))) __bf16 v16bf;
typedef __attribute__((ext_vector_type(8)))  __bf16 v8bf;
typedef __attribute__((ext_vector_type(8)))  float  v8f;

// ---------------------------------------------------------------------------
// Implicit-GEMM conv, bf16 activations & weights, f32 accumulate.
// conv_fast: one wave32 computes a 32(M voxels) x 32(N cout) tile with
//   4 accumulators; per 32-K step: 4 bf16 A loads? no -- 2 A-subtile loads +
//   2 B-subtile loads (8x global_load_b128 total) feed 4 WMMAs (2 loads/WMMA).
// Branchless OOB handling via pointer-select to a zeroed line.
// Fused epilogue: BN-affine, residual add, ReLU, active-site mask.
// ---------------------------------------------------------------------------
struct ConvP {
  const __bf16* in;          // bf16 activations (fast path)
  const float*  inF;         // f32 input (generic path: stem.0 reads x)
  const __bf16* wT;          // [slice][Npad][K32] bf16, zero padded
  const float* scale;
  const float* bias;
  const __bf16* residual;
  const unsigned char* mask;
  void* out;                 // bf16, or f32 if outF32
  const float* zerobuf;      // >=32B of zeros
  int Din, Dout, Cin, Cout;
  int ks, stride, pad;
  int Ktot, K32;
  int Nvox;                  // M extent (Dout^3, or Din^3 per parity if trans)
  int trans;                 // transposed k2s2: parity = blockIdx.z
  int relu, affine, outF32;
  int wsliceStride;          // Npad*K32 for trans slices
  int cinShift;              // log2(Cin) for the generic kernel
};

__device__ __forceinline__ v16bf mk16(v8bf lo, v8bf hi)
{
  v16bf r;
  #pragma unroll
  for (int j = 0; j < 8; ++j) { r[j] = lo[j]; r[j + 8] = hi[j]; }
  return r;
}

__device__ __forceinline__ void conv_epi(const ConvP& p, const v8f& acc,
    int voxBase, int half, int ncol, int Dm, int tz, int ty, int tx)
{
  if (ncol >= p.Cout) return;
  const float sc = p.affine ? p.scale[ncol] : 1.0f;
  const float bo = p.affine ? p.bias[ncol]  : 0.0f;
  #pragma unroll
  for (int r = 0; r < 8; ++r) {
    int voxm = voxBase + r + 8 * half;    // C/D layout: VGPR r -> M=r+8*half
    if (voxm >= p.Nvox) continue;
    int midx;
    if (p.trans) {
      int oz = voxm / (Dm * Dm);
      int rr = voxm - oz * Dm * Dm;
      int oy = rr / Dm;
      int ox = rr - oy * Dm;
      midx = ((2 * oz + tz) * p.Dout + (2 * oy + ty)) * p.Dout + (2 * ox + tx);
    } else {
      midx = voxm;
    }
    long oidx = (long)midx * p.Cout + ncol;
    float y = acc[r] * sc + bo;
    if (p.residual) y += (float)p.residual[oidx];
    if (p.relu)     y = fmaxf(y, 0.0f);
    if (p.mask && !p.mask[midx]) y = 0.0f;
    if (p.outF32) ((float*)p.out)[oidx] = y;
    else          ((__bf16*)p.out)[oidx] = (__bf16)y;
  }
}

// Fast path: requires Cin % 32 == 0 (all layers except stem.0).
__global__ __launch_bounds__(32) void conv_fast(ConvP p)
{
  const int lane = threadIdx.x;
  const int half = lane >> 4;
  const int mr   = lane & 15;
  const int tileM = blockIdx.x;            // 32 voxel rows
  const int tileN = blockIdx.y;            // 32 cout cols
  const int par   = blockIdx.z;
  const int tz = (par >> 2) & 1, ty = (par >> 1) & 1, tx = par & 1;
  const int flip = p.trans ? (((1 - tz) * 2 + (1 - ty)) * 2 + (1 - tx)) : 0;

  const int ncol0 = tileN * 32 + mr;       // < Npad (padded to 32)
  const int ncol1 = ncol0 + 16;
  const __bf16* wS    = p.wT + (size_t)flip * p.wsliceStride + 16 * half;
  const __bf16* wrow0 = wS + (size_t)ncol0 * p.K32;
  const __bf16* wrow1 = wS + (size_t)ncol1 * p.K32;

  const int Dm = p.trans ? p.Din : p.Dout;
  const int vox0 = tileM * 32 + mr;
  const int vox1 = vox0 + 16;
  const bool mv0 = vox0 < p.Nvox;
  const bool mv1 = vox1 < p.Nvox;
  int vz0, vy0, vx0, vz1, vy1, vx1;
  {
    int t = mv0 ? vox0 : 0;
    vz0 = t / (Dm * Dm); int r = t - vz0 * Dm * Dm; vy0 = r / Dm; vx0 = r - vy0 * Dm;
    t = mv1 ? vox1 : 0;
    vz1 = t / (Dm * Dm); r = t - vz1 * Dm * Dm; vy1 = r / Dm; vx1 = r - vy1 * Dm;
  }

  v8f acc00, acc01, acc10, acc11;
  #pragma unroll
  for (int i = 0; i < 8; ++i) { acc00[i] = 0.f; acc01[i] = 0.f; acc10[i] = 0.f; acc11[i] = 0.f; }

  const __bf16* zbf = (const __bf16*)p.zerobuf;
  const int taps = p.trans ? 1 : p.ks * p.ks * p.ks;
  int dz = 0, dy = 0, dx = 0;
  for (int tap = 0; tap < taps; ++tap) {
    int iz0, iy0, ix0, iz1, iy1, ix1;
    bool v0, v1;
    if (p.trans) {
      iz0 = vz0; iy0 = vy0; ix0 = vx0; v0 = mv0;
      iz1 = vz1; iy1 = vy1; ix1 = vx1; v1 = mv1;
    } else {
      iz0 = vz0 * p.stride + dz - p.pad; iy0 = vy0 * p.stride + dy - p.pad;
      ix0 = vx0 * p.stride + dx - p.pad;
      iz1 = vz1 * p.stride + dz - p.pad; iy1 = vy1 * p.stride + dy - p.pad;
      ix1 = vx1 * p.stride + dx - p.pad;
      v0 = mv0 && iz0 >= 0 && iz0 < p.Din && iy0 >= 0 && iy0 < p.Din &&
           ix0 >= 0 && ix0 < p.Din;
      v1 = mv1 && iz1 >= 0 && iz1 < p.Din && iy1 >= 0 && iy1 < p.Din &&
           ix1 >= 0 && ix1 < p.Din;
      if (++dx == p.ks) { dx = 0; if (++dy == p.ks) { dy = 0; ++dz; } }
    }
    const __bf16* ab0 = p.in + ((size_t)((iz0 * p.Din + iy0) * p.Din + ix0)) * p.Cin;
    const __bf16* ab1 = p.in + ((size_t)((iz1 * p.Din + iy1) * p.Din + ix1)) * p.Cin;
    const __bf16* bb0 = wrow0 + tap * p.Cin;
    const __bf16* bb1 = wrow1 + tap * p.Cin;

    for (int cb = 0; cb < p.Cin; cb += 32) {
      // A fragments: two 8-elem runs per lane (K = 16h + j), 16B each.
      const v8bf* a0l = (const v8bf*)(v0 ? ab0 + cb + 16 * half      : zbf);
      const v8bf* a0h = (const v8bf*)(v0 ? ab0 + cb + 16 * half + 8  : zbf);
      const v8bf* a1l = (const v8bf*)(v1 ? ab1 + cb + 16 * half      : zbf);
      const v8bf* a1h = (const v8bf*)(v1 ? ab1 + cb + 16 * half + 8  : zbf);
      v16bf a0 = mk16(a0l[0], a0h[0]);
      v16bf a1 = mk16(a1l[0], a1h[0]);
      // B fragments: 16 consecutive bf16 per lane (K = 16h + j), padded.
      const v8bf* b0p = (const v8bf*)(bb0 + cb);
      const v8bf* b1p = (const v8bf*)(bb1 + cb);
      v16bf b0 = mk16(b0p[0], b0p[1]);
      v16bf b1 = mk16(b1p[0], b1p[1]);

      acc00 = __builtin_amdgcn_wmma_f32_16x16x32_bf16(false, a0, false, b0, (short)0, acc00, false, false);
      acc01 = __builtin_amdgcn_wmma_f32_16x16x32_bf16(false, a0, false, b1, (short)0, acc01, false, false);
      acc10 = __builtin_amdgcn_wmma_f32_16x16x32_bf16(false, a1, false, b0, (short)0, acc10, false, false);
      acc11 = __builtin_amdgcn_wmma_f32_16x16x32_bf16(false, a1, false, b1, (short)0, acc11, false, false);
    }
  }
  conv_epi(p, acc00, tileM * 32,      half, ncol0, Dm, tz, ty, tx);
  conv_epi(p, acc01, tileM * 32,      half, ncol1, Dm, tz, ty, tx);
  conv_epi(p, acc10, tileM * 32 + 16, half, ncol0, Dm, tz, ty, tx);
  conv_epi(p, acc11, tileM * 32 + 16, half, ncol1, Dm, tz, ty, tx);
}

// Generic path (stem.0 only): f32 input, power-of-two Cin, ks==3 s1 p1,
// per-element A gather (branchless pointer select), 16x16 tile.
__global__ __launch_bounds__(32) void conv_gen(ConvP p)
{
  const int lane = threadIdx.x;
  const int half = lane >> 4;
  const int mr   = lane & 15;
  const int tileM = blockIdx.x;
  const int tileN = blockIdx.y;

  const int ncol = tileN * 16 + mr;
  const __bf16* wrow = p.wT + (size_t)ncol * p.K32 + 16 * half;

  const int  voxA   = tileM * 16 + mr;
  const bool mvalid = voxA < p.Nvox;
  const int  Dm     = p.Dout;
  int vz, vy, vx;
  {
    int t = mvalid ? voxA : 0;
    vz = t / (Dm * Dm);
    int r = t - vz * Dm * Dm;
    vy = r / Dm;
    vx = r - vy * Dm;
  }

  v8f acc;
  #pragma unroll
  for (int i = 0; i < 8; ++i) acc[i] = 0.0f;

  const int cmask = p.Cin - 1;
  for (int kb = 0; kb < p.K32; kb += 32) {
    v16bf af, bf;
    #pragma unroll
    for (int j = 0; j < 16; ++j) {
      int ka  = kb + 16 * half + j;
      int tap = ka >> p.cinShift;
      int c   = ka & cmask;
      int dz  = tap / 9;
      int rr  = tap - dz * 9;
      int dy  = rr / 3;
      int dx  = rr - dy * 3;
      int iz  = vz + dz - 1;
      int iy  = vy + dy - 1;
      int ix  = vx + dx - 1;
      bool ok = mvalid && ka < p.Ktot && iz >= 0 && iz < p.Din &&
                iy >= 0 && iy < p.Din && ix >= 0 && ix < p.Din;
      const float* ap = ok
          ? (p.inF + ((size_t)((iz * p.Din + iy) * p.Din + ix)) * p.Cin + c)
          : p.zerobuf;
      af[j] = (__bf16)(*ap);
    }
    const v8bf* bp = (const v8bf*)(wrow + kb);
    bf = mk16(bp[0], bp[1]);
    acc = __builtin_amdgcn_wmma_f32_16x16x32_bf16(
        false, af, false, bf, (short)0, acc, false, false);
  }
  conv_epi(p, acc, tileM * 16, half, ncol, Dm, 0, 0, 0);
}

// Weight prep: wT[slice][n][k] = bf16(w[slice][k][n]), zero-padded to K32/Npad.
__global__ void wtrans_kernel(const float* w, __bf16* wT, int Ktot, int K32,
                              int Cout, long sInStride, long sOutStride)
{
  int k = blockIdx.x * 256 + threadIdx.x;
  if (k >= K32) return;
  int n = blockIdx.y;
  int s = blockIdx.z;
  float v = (k < Ktot && n < Cout) ? w[s * sInStride + (long)k * Cout + n] : 0.0f;
  wT[s * sOutStride + (long)n * K32 + k] = (__bf16)v;
}

// Channel concat (bf16): out[v, 0:Ca]=a, out[v, Ca:Ca+Cb]=b
__global__ void concat2_kernel(const unsigned short* a, int Ca,
                               const unsigned short* b, int Cb,
                               unsigned short* out, int total)
{
  int i = blockIdx.x * blockDim.x + threadIdx.x;
  if (i >= total) return;
  int C = Ca + Cb;
  int v = i / C;
  int c = i - v * C;
  out[i] = (c < Ca) ? a[(long)v * Ca + c] : b[(long)v * Cb + (c - Ca)];
}

// ---------------------------------------------------------------------------
// Host side
// ---------------------------------------------------------------------------
namespace {

struct CB { const float *w, *s, *b; };

struct Cursor {
  void* const* d_in;
  bool  flat;
  int   idx;
  const float* base;
  long  off;
  const float* next(long n) {
    if (flat) { const float* p = base + off; off += n; return p; }
    return (const float*)d_in[idx++];
  }
  CB cb(int k, long cin, long cout) {
    CB r;
    r.w = next((long)k * k * k * cin * cout);
    r.s = next(cout);
    r.b = next(cout);
    return r;
  }
};

} // namespace

extern "C" void kernel_launch(void* const* d_in, const int* in_sizes, int n_in,
                              void* d_out, int out_size, void* d_ws, size_t ws_size,
                              hipStream_t stream)
{
  (void)in_sizes; (void)out_size; (void)ws_size;
  const float* x = (const float*)d_in[0];
  const unsigned char* mk[5];
  for (int i = 0; i < 5; ++i) mk[i] = (const unsigned char*)d_in[1 + i];

  Cursor cur;
  cur.flat = (n_in < 158);            // 6 tensors + 152 leaves if separate
  cur.d_in = d_in;
  cur.idx  = 6;
  cur.base = (const float*)d_in[6];
  cur.off  = 0;

  // ---- workspace (byte bump allocator; deterministic per call) ----
  char* wsb = (char*)d_ws;
  size_t wo = 0;
  auto alloc = [&](size_t bytes) {
    void* p = wsb + wo; wo += (bytes + 63) & ~(size_t)63; return p;
  };
  float*  zb = (float*)alloc(256);               // zero line for OOB gathers
  __bf16* s0 = (__bf16*)alloc(32768UL * 32 * 2);
  __bf16* s1 = (__bf16*)alloc(4096UL * 32 * 2);
  __bf16* s2 = (__bf16*)alloc(512UL * 64 * 2);
  __bf16* s3 = (__bf16*)alloc(64UL * 128 * 2);
  __bf16* s4 = (__bf16*)alloc(8UL * 256 * 2);
  __bf16* BA = (__bf16*)alloc(32768UL * 128 * 2);
  __bf16* BB = (__bf16*)alloc(32768UL * 128 * 2);
  __bf16* BC = (__bf16*)alloc(32768UL * 128 * 2);

  hipMemsetAsync(zb, 0, 256, stream);

  auto conv_launch = [&](const void* in, const CB& cb, const __bf16* res,
                         const unsigned char* msk, void* out,
                         int Din, int Dout, int Cin, int Cout,
                         int ks, int stride, int pad, int trans, int relu,
                         int affine, int outF32) {
    const int slices = trans ? 8 : 1;
    const int ktot   = trans ? Cin : ks * ks * ks * Cin;
    const int K32    = (ktot + 31) & ~31;
    const int Npad   = (Cout + 31) & ~31;
    const long per   = (long)Npad * K32;
    __bf16* wT = (__bf16*)alloc((size_t)per * slices * 2);
    {
      dim3 g((K32 + 255) / 256, Npad, slices);
      wtrans_kernel<<<g, dim3(256), 0, stream>>>(
          cb.w, wT, ktot, K32, Cout, (long)Cin * Cout, per);
    }
    ConvP p;
    p.in = (const __bf16*)in; p.inF = (const float*)in;
    p.wT = wT; p.scale = cb.s; p.bias = cb.b;
    p.residual = res; p.mask = msk; p.out = out; p.zerobuf = zb;
    p.Din = Din; p.Dout = Dout; p.Cin = Cin; p.Cout = Cout;
    p.ks = ks; p.stride = stride; p.pad = pad;
    p.Ktot = ktot; p.K32 = K32;
    p.Nvox = trans ? Din * Din * Din : Dout * Dout * Dout;
    p.trans = trans; p.relu = relu; p.affine = affine; p.outF32 = outF32;
    p.wsliceStride = (int)per;
    p.cinShift = __builtin_ctz((unsigned)Cin);
    if (Cin % 32 == 0) {
      dim3 grid((p.Nvox + 31) / 32, Npad / 32, slices);
      conv_fast<<<grid, dim3(32), 0, stream>>>(p);
    } else {
      dim3 grid((p.Nvox + 15) / 16, Npad / 16, 1);
      conv_gen<<<grid, dim3(32), 0, stream>>>(p);
    }
  };

  const int NCv[9] = {32, 32, 64, 128, 256, 256, 128, 96, 96};
  const int Dl[5]  = {32, 16, 8, 4, 2};
  __bf16* skips[5] = {s0, s1, s2, s3, s4};

  // ---- stem ----
  CB st0 = cur.cb(3, 4, 32);
  CB st1 = cur.cb(3, 32, 32);
  conv_launch(x,  st0, nullptr, mk[0], BA, 32, 32, 4, 32, 3, 1, 1, 0, 1, 1, 0);
  conv_launch(BA, st1, nullptr, mk[0], s0, 32, 32, 32, 32, 3, 1, 1, 0, 1, 1, 0);

  // ---- encoder ----
  for (int k = 0; k < 4; ++k) {
    int cin = NCv[k], cout = NCv[k + 1];
    CB dn = cur.cb(2, cin, cin);
    CB c1 = cur.cb(3, cin, cout);
    CB c2 = cur.cb(3, cout, cout);
    bool has_sc = (cin != cout);
    CB scb{nullptr, nullptr, nullptr};
    if (has_sc) scb = cur.cb(1, cin, cout);
    CB d1 = cur.cb(3, cout, cout);
    CB d2 = cur.cb(3, cout, cout);
    int Di = Dl[k], Do = Dl[k + 1];
    const unsigned char* m = mk[k + 1];

    conv_launch(skips[k], dn, nullptr, m, BA, Di, Do, cin, cin, 2, 2, 0, 0, 1, 1, 0);
    conv_launch(BA, c1, nullptr, m, BB, Do, Do, cin, cout, 3, 1, 1, 0, 1, 1, 0);
    __bf16* hout;
    if (has_sc) {
      conv_launch(BA, scb, nullptr, m, BC, Do, Do, cin, cout, 1, 1, 0, 0, 0, 1, 0);
      conv_launch(BB, c2, BC, m, BA, Do, Do, cout, cout, 3, 1, 1, 0, 1, 1, 0);
      hout = BA;
    } else {
      conv_launch(BB, c2, BA, m, BC, Do, Do, cout, cout, 3, 1, 1, 0, 1, 1, 0);
      hout = BC;
    }
    __bf16* t = (hout == BA) ? BB : BA;
    conv_launch(hout, d1, nullptr, m, t, Do, Do, cout, cout, 3, 1, 1, 0, 1, 1, 0);
    conv_launch(t, d2, hout, m, skips[k + 1], Do, Do, cout, cout, 3, 1, 1, 0, 1, 1, 0);
  }

  // ---- decoder ----
  const __bf16* y = s4;
  __bf16* b0 = BA; __bf16* b1 = BB; __bf16* b2 = BC;
  for (int k = 0; k < 4; ++k) {
    int cup_in = NCv[k + 4], cout = NCv[k + 5];
    int cskip  = NCv[3 - k];
    int cin    = cout + cskip;
    int lvl    = 3 - k;
    int Di = Dl[lvl + 1], Do = Dl[lvl];
    const unsigned char* m = mk[lvl];

    CB upw = cur.cb(2, cup_in, cout);
    CB c1  = cur.cb(3, cin, cout);
    CB c2  = cur.cb(3, cout, cout);
    CB scb = cur.cb(1, cin, cout);
    CB d1  = cur.cb(3, cout, cout);
    CB d2  = cur.cb(3, cout, cout);

    conv_launch(y, upw, nullptr, m, b0, Di, Do, cup_in, cout, 2, 2, 0, 1, 1, 1, 0);
    int Nv = Do * Do * Do;
    int tot = Nv * cin;
    concat2_kernel<<<dim3((tot + 255) / 256), dim3(256), 0, stream>>>(
        (const unsigned short*)b0, cout, (const unsigned short*)skips[lvl],
        cskip, (unsigned short*)b1, tot);
    conv_launch(b1, c1, nullptr, m, b2, Do, Do, cin, cout, 3, 1, 1, 0, 1, 1, 0);
    conv_launch(b1, scb, nullptr, m, b0, Do, Do, cin, cout, 1, 1, 0, 0, 0, 1, 0);
    conv_launch(b2, c2, b0, m, b1, Do, Do, cout, cout, 3, 1, 1, 0, 1, 1, 0);
    conv_launch(b1, d1, nullptr, m, b2, Do, Do, cout, cout, 3, 1, 1, 0, 1, 1, 0);
    conv_launch(b2, d2, b1, m, b0, Do, Do, cout, cout, 3, 1, 1, 0, 1, 1, 0);
    y = b0;
    __bf16* t0 = b0; b0 = b1; b1 = b2; b2 = t0;
  }

  // ---- heads (f32 output into d_out) ----
  const int co[5] = {3, 4, 6, 9, 15};
  long off = 0;
  for (int i = 0; i < 5; ++i) {
    CB hb;
    hb.w = cur.next(96L * co[i]);
    hb.s = nullptr; hb.b = nullptr;
    conv_launch(y, hb, nullptr, mk[0], (float*)d_out + off,
                32, 32, 96, co[i], 1, 1, 0, 0, 0, 0, 1);
    off += 32768L * co[i];
  }
}
